// TransformerModel_51049981281253
// MI455X (gfx1250) — matrix operand order repaired and checked by reading, not compiled, not run on hardware
//
#include <hip/hip_runtime.h>
#include <math.h>

#define DEVI __device__ __forceinline__

typedef _Float16 f16;
typedef __attribute__((ext_vector_type(16))) _Float16 v16h;
typedef __attribute__((ext_vector_type(8)))  _Float16 h8;
typedef __attribute__((ext_vector_type(2)))  _Float16 h2;
typedef __attribute__((ext_vector_type(8)))  float    v8f;
typedef __attribute__((ext_vector_type(4)))  float    f32x4;
typedef __attribute__((ext_vector_type(4)))  int      i32x4;

#define AS1 __attribute__((address_space(1)))
#define AS3 __attribute__((address_space(3)))

#if defined(__has_builtin)
#if __has_builtin(__builtin_amdgcn_global_load_async_to_lds_b128)
#define USE_ASYNC_LDS 1
#else
#define USE_ASYNC_LDS 0
#endif
#else
#define USE_ASYNC_LDS 0
#endif

union AFrag { v16h v; h8 h[2]; };

DEVI v8f wmma_f16(v16h a, v16h b, v8f c) {
  // D = A(16x32 f16) * B(32x16 f16) + C(16x16 f32)
  return __builtin_amdgcn_wmma_f32_16x16x32_f16(false, a, false, b, (short)0, c,
                                                false, false);
}

DEVI v8f vzero() {
  v8f z;
#pragma unroll
  for (int i = 0; i < 8; i++) z[i] = 0.0f;
  return z;
}

// 16-byte global -> LDS copy; async DMA path on CDNA5 (ASYNCcnt-tracked).
DEVI void copy16_g2l(const f16* g, f16* l) {
#if USE_ASYNC_LDS
  __builtin_amdgcn_global_load_async_to_lds_b128((AS1 i32x4*)g, (AS3 i32x4*)l,
                                                 0, 0);
#else
  *(h8*)l = *(const h8*)g;
#endif
}

DEVI void wait_async() {
#if USE_ASYNC_LDS
#if __has_builtin(__builtin_amdgcn_s_wait_asynccnt)
  __builtin_amdgcn_s_wait_asynccnt(0);
#else
  asm volatile("s_wait_asynccnt 0x0" ::: "memory");
#endif
  asm volatile("" ::: "memory");
#endif
}

// ---------------------------------------------------------------------------
// Embedding: x[row,:] = tok_emb[tokens[row],:] + pos_emb[row%S,:]
// ---------------------------------------------------------------------------
__global__ __launch_bounds__(256) void k_embed(const int* __restrict__ tok,
                                               const float* __restrict__ te,
                                               const float* __restrict__ pe,
                                               float* __restrict__ x) {
  const int row = blockIdx.x;       // 0..2047 (b*1024+s)
  const int s   = row & 1023;
  const int tk  = tok[row];
  const int c   = threadIdx.x * 4;
  f32x4 a = *(const f32x4*)(te + (size_t)tk * 1024 + c);
  f32x4 b = *(const f32x4*)(pe + (size_t)s * 1024 + c);
  *(f32x4*)(x + (size_t)row * 1024 + c) = a + b;
}

// ---------------------------------------------------------------------------
// LayerNorm over H=1024, writes f16 (GEMM A input)
// ---------------------------------------------------------------------------
__global__ __launch_bounds__(256) void k_ln(const float* __restrict__ x,
                                            const float* __restrict__ g,
                                            const float* __restrict__ b,
                                            f16* __restrict__ y) {
  const int row = blockIdx.x;
  const int t   = threadIdx.x;
  f32x4 v = *(const f32x4*)(x + (size_t)row * 1024 + t * 4);
  float s1 = v[0] + v[1] + v[2] + v[3];
  float s2 = v[0]*v[0] + v[1]*v[1] + v[2]*v[2] + v[3]*v[3];
#pragma unroll
  for (int off = 16; off; off >>= 1) {
    s1 += __shfl_down(s1, off);
    s2 += __shfl_down(s2, off);
  }
  __shared__ float w1[8], w2[8];
  const int wid = t >> 5, lane = t & 31;
  if (lane == 0) { w1[wid] = s1; w2[wid] = s2; }
  __syncthreads();
  float S1 = 0.f, S2 = 0.f;
#pragma unroll
  for (int i = 0; i < 8; i++) { S1 += w1[i]; S2 += w2[i]; }
  const float mean = S1 * (1.0f / 1024.0f);
  const float var  = S2 * (1.0f / 1024.0f) - mean * mean;
  const float rs   = rsqrtf(var + 1e-5f);
#pragma unroll
  for (int i = 0; i < 4; i++) {
    const int c = t * 4 + i;
    y[(size_t)row * 1024 + c] = (f16)((v[i] - mean) * rs * g[c] + b[c]);
  }
}

// ---------------------------------------------------------------------------
// WMMA GEMM: C[M,N] = A[M,K](f16) * B[K,N](f32 -> f16 in LDS)
// 256 threads / 8 waves, block tile 128x128, K-step 32, wave tile 64x32.
// Double-buffered LDS; A staged via async global->LDS DMA, B staged through
// registers (needs f32->f16 convert + transpose).
// EPI: 0 = f16 store, 1 = GELU(exact)+f16 store, 2 = Cf32 += (residual add),
//      3 = f32 store.
// ---------------------------------------------------------------------------
template <int EPI>
__global__ __launch_bounds__(256) void k_gemm(const f16* __restrict__ A,
                                              const float* __restrict__ B,
                                              f16* __restrict__ Cf16,
                                              float* __restrict__ Cf32,
                                              const int M, const int N,
                                              const int K) {
  __shared__ f16 As[2][128][40];   // [m][k], padded
  __shared__ f16 Bs[2][128][40];   // transposed: [n][k], padded
  const int t    = threadIdx.x;
  const int wid  = t >> 5, lane = t & 31;
  const int hl   = lane >> 4;         // half-wave: 0 | 1
  const int l15  = lane & 15;
  const int m0   = blockIdx.y * 128, n0 = blockIdx.x * 128;
  const int wm   = wid >> 2, wn = wid & 3;

  v8f acc[4][2];
#pragma unroll
  for (int i = 0; i < 4; i++)
#pragma unroll
    for (int j = 0; j < 2; j++) acc[i][j] = vzero();

  const int arow = t >> 1, ac0 = (t & 1) * 16;       // A loader: 128 rows x 32
  const int bkp  = (t >> 4) * 2, bn0 = (t & 15) * 8; // B loader: 32k x 128n

  f32x4 r0a, r0b, r1a, r1b;  // staged B registers

  auto asyncA = [&](int kk, int buf) {
    const f16* ap = A + (size_t)(m0 + arow) * K + kk + ac0;
    copy16_g2l(ap,     &As[buf][arow][ac0]);
    copy16_g2l(ap + 8, &As[buf][arow][ac0 + 8]);
  };
  auto loadB = [&](int kk) {
    const float* bp0 = B + (size_t)(kk + bkp) * N + n0 + bn0;
    const float* bp1 = bp0 + N;
    r0a = *(const f32x4*)bp0;  r0b = *(const f32x4*)(bp0 + 4);
    r1a = *(const f32x4*)bp1;  r1b = *(const f32x4*)(bp1 + 4);
  };
  auto storeB = [&](int buf) {
#pragma unroll
    for (int j = 0; j < 4; j++) {
      h2 p; p[0] = (f16)r0a[j]; p[1] = (f16)r1a[j];
      *(h2*)&Bs[buf][bn0 + j][bkp] = p;
      h2 q; q[0] = (f16)r0b[j]; q[1] = (f16)r1b[j];
      *(h2*)&Bs[buf][bn0 + 4 + j][bkp] = q;
    }
  };

  // Pipeline prologue: stage K-step 0 into buffer 0.
  asyncA(0, 0);
  loadB(0);
  storeB(0);
  wait_async();
  __syncthreads();

  int buf = 0;
  for (int kk = 0; kk < K; kk += 32) {
    const bool more = (kk + 32) < K;
    if (more) {              // issue next stage's global traffic first
      asyncA(kk + 32, buf ^ 1);
      loadB(kk + 32);
    }
    // --- fragments + WMMA from LDS[buf] ---
    AFrag af[4], bf[2];
    const int akb = hl * 8;  // A layout: halves K {0..7,16..23} / {8..15,24..31}
#pragma unroll
    for (int mf = 0; mf < 4; mf++) {
      const f16* p = &As[buf][wm * 64 + mf * 16 + l15][akb];
      af[mf].h[0] = *(const h8*)p;
      af[mf].h[1] = *(const h8*)(p + 16);
    }
    const int bkb = hl * 8;  // B layout: halves K {0..7,16..23} / {8..15,24..31}
#pragma unroll
    for (int nf = 0; nf < 2; nf++) {
      const f16* p = &Bs[buf][wn * 32 + nf * 16 + l15][bkb];
      bf[nf].h[0] = *(const h8*)p;
      bf[nf].h[1] = *(const h8*)(p + 16);
    }
#pragma unroll
    for (int mf = 0; mf < 4; mf++)
#pragma unroll
      for (int nf = 0; nf < 2; nf++)
        acc[mf][nf] = wmma_f16(af[mf].v, bf[nf].v, acc[mf][nf]);
    if (more) storeB(buf ^ 1);
    wait_async();
    __syncthreads();
    buf ^= 1;
  }

  // --- epilogue (C layout: vgpr r -> row r + 8*hl, lane l15 -> col) ---
#pragma unroll
  for (int mf = 0; mf < 4; mf++)
#pragma unroll
    for (int nf = 0; nf < 2; nf++)
#pragma unroll
      for (int r = 0; r < 8; r++) {
        const int m = m0 + wm * 64 + mf * 16 + r + hl * 8;
        const int n = n0 + wn * 32 + nf * 16 + l15;
        float val = acc[mf][nf][r];
        const size_t idx = (size_t)m * N + n;
        if (EPI == 0) {
          Cf16[idx] = (f16)val;
        } else if (EPI == 1) {
          val = 0.5f * val * (1.0f + erff(val * 0.70710678118654752f));
          Cf16[idx] = (f16)val;
        } else if (EPI == 2) {
          Cf32[idx] += val;
        } else {
          Cf32[idx] = val;
        }
      }
}

// ---------------------------------------------------------------------------
// Flash attention (causal). qkv: f16 [B*S, 3H], rows b*1024+s; Q at col h*64,
// K at 1024+h*64, V at 2048+h*64. Block = (128 q rows, head, batch), 8 waves,
// each wave owns 16 q rows. o: f16 [B*S, H].
// ---------------------------------------------------------------------------
__global__ __launch_bounds__(256) void k_attn(const f16* __restrict__ qkv,
                                              f16* __restrict__ o) {
  __shared__ f16 Ks[64][72];       // [key][d]
  __shared__ f16 Vt[64][72];       // [d][key] (transposed)
  __shared__ f16 Ps[8][16][72];    // per-wave P round-trip (C->A layout)
  const int t = threadIdx.x, wid = t >> 5, lane = t & 31;
  const int hl = lane >> 4, l15 = lane & 15;
  const int qblk = blockIdx.x, h = blockIdx.y, b = blockIdx.z;
  const int qbase = qblk * 128;
  const int wq0 = qbase + wid * 16;
  const int RS = 3 * 1024;

  // Q fragments (persist in registers across all k-tiles)
  AFrag qa[2];
  {
    const f16* qp0 = qkv + (size_t)(b * 1024 + wq0 + l15) * RS + h * 64;
    const int kb = hl * 8;
#pragma unroll
    for (int ks = 0; ks < 2; ks++) {
      const f16* p = qp0 + ks * 32 + kb;
      qa[ks].h[0] = *(const h8*)p;
      qa[ks].h[1] = *(const h8*)(p + 16);
    }
  }

  v8f oa[4];
#pragma unroll
  for (int f = 0; f < 4; f++) oa[f] = vzero();
  float mrow[8], lrow[8];
#pragma unroll
  for (int r = 0; r < 8; r++) { mrow[r] = -1e30f; lrow[r] = 0.0f; }

  const int ldkey = t >> 2, ldd = (t & 3) * 16;  // K/V tile loaders
  const int bkb = hl * 8;
  const int ktmax = qblk * 2 + 1;

  for (int kt = 0; kt <= ktmax; kt++) {
    const int kk0 = kt * 64;
    // --- stage K (async straight copy) and V (transposed, manual) tiles ---
    {
      const f16* kp = qkv + (size_t)(b * 1024 + kk0 + ldkey) * RS + 1024 + h * 64 + ldd;
      copy16_g2l(kp,     &Ks[ldkey][ldd]);
      copy16_g2l(kp + 8, &Ks[ldkey][ldd + 8]);
      const f16* vp = kp + 1024;
      h8 v0 = *(const h8*)vp, v1 = *(const h8*)(vp + 8);
#pragma unroll
      for (int i = 0; i < 8; i++) {
        Vt[ldd + i][ldkey]     = v0[i];
        Vt[ldd + 8 + i][ldkey] = v1[i];
      }
    }
    wait_async();
    __syncthreads();

    if (kk0 <= wq0 + 15) {   // wave-uniform: EXEC stays all-ones for WMMA
      // --- S = Q K^T over 64 keys (4 n-frags x 2 k-steps of 32) ---
      v8f sf[4];
#pragma unroll
      for (int nf = 0; nf < 4; nf++) {
        AFrag b0, b1;
        const f16* p = &Ks[nf * 16 + l15][bkb];
        b0.h[0] = *(const h8*)p;        b0.h[1] = *(const h8*)(p + 16);
        b1.h[0] = *(const h8*)(p + 32); b1.h[1] = *(const h8*)(p + 48);
        sf[nf] = wmma_f16(qa[0].v, b0.v, vzero());
        sf[nf] = wmma_f16(qa[1].v, b1.v, sf[nf]);
      }
      // --- scale, causal mask, online softmax (rows live per-VGPR) ---
#pragma unroll
      for (int r = 0; r < 8; r++) {
        const int q = wq0 + r + hl * 8;
        float mx = -1e30f;
#pragma unroll
        for (int nf = 0; nf < 4; nf++) {
          const int key = kk0 + nf * 16 + l15;
          float val = sf[nf][r] * 0.125f;          // 1/sqrt(64)
          val = (key <= q) ? val : -1e30f;
          sf[nf][r] = val;
          mx = fmaxf(mx, val);
        }
        mx = fmaxf(mx, __shfl_xor(mx, 1));
        mx = fmaxf(mx, __shfl_xor(mx, 2));
        mx = fmaxf(mx, __shfl_xor(mx, 4));
        mx = fmaxf(mx, __shfl_xor(mx, 8));
        const float mnew = fmaxf(mrow[r], mx);
        const float alpha = __expf(mrow[r] - mnew);
        lrow[r] *= alpha;
        mrow[r] = mnew;
#pragma unroll
        for (int f = 0; f < 4; f++) oa[f][r] *= alpha;
        float rsum = 0.0f;
#pragma unroll
        for (int nf = 0; nf < 4; nf++) {
          const float pv = __expf(sf[nf][r] - mnew);
          rsum += pv;
          Ps[wid][r + hl * 8][nf * 16 + l15] = (f16)pv;
        }
        rsum += __shfl_xor(rsum, 1);
        rsum += __shfl_xor(rsum, 2);
        rsum += __shfl_xor(rsum, 4);
        rsum += __shfl_xor(rsum, 8);
        lrow[r] += rsum;
      }
      // same-wave cross-lane LDS RAW: wait DS stores before A-frag reload
      asm volatile("s_wait_dscnt 0x0" ::: "memory");
      // --- O += P V ---
      const int akb = hl * 8;
#pragma unroll
      for (int ks = 0; ks < 2; ks++) {
        AFrag pa;
        const f16* pp = &Ps[wid][l15][ks * 32 + akb];
        pa.h[0] = *(const h8*)pp;
        pa.h[1] = *(const h8*)(pp + 16);
#pragma unroll
        for (int f = 0; f < 4; f++) {
          AFrag vb;
          const f16* vp2 = &Vt[f * 16 + l15][ks * 32 + bkb];
          vb.h[0] = *(const h8*)vp2;
          vb.h[1] = *(const h8*)(vp2 + 16);
          oa[f] = wmma_f16(pa.v, vb.v, oa[f]);
        }
      }
    }
    __syncthreads();
  }

  // --- finalize: O /= l, write f16 ---
#pragma unroll
  for (int r = 0; r < 8; r++) {
    const float inv = 1.0f / lrow[r];
    const int row = b * 1024 + wq0 + r + hl * 8;
#pragma unroll
    for (int f = 0; f < 4; f++)
      o[(size_t)row * 1024 + h * 64 + f * 16 + l15] = (f16)(oa[f][r] * inv);
  }
}

// ---------------------------------------------------------------------------
// Driver
// ---------------------------------------------------------------------------
extern "C" void kernel_launch(void* const* d_in, const int* in_sizes, int n_in,
                              void* d_out, int out_size, void* d_ws,
                              size_t ws_size, hipStream_t stream) {
  (void)in_sizes; (void)n_in; (void)out_size; (void)ws_size;
  const int*   tokens  = (const int*)d_in[0];
  const float* tok_emb = (const float*)d_in[1];
  const float* pos_emb = (const float*)d_in[2];
  const float* ln1_g   = (const float*)d_in[3];
  const float* ln1_b   = (const float*)d_in[4];
  const float* ln2_g   = (const float*)d_in[5];
  const float* ln2_b   = (const float*)d_in[6];
  const float* attn_w  = (const float*)d_in[7];
  const float* attn_pw = (const float*)d_in[8];
  const float* fc_w    = (const float*)d_in[9];
  const float* proj_w  = (const float*)d_in[10];
  const float* lnf_g   = (const float*)d_in[11];
  const float* lnf_b   = (const float*)d_in[12];
  const float* head_w  = (const float*)d_in[13];
  float* out = (float*)d_out;

  char* ws = (char*)d_ws;
  float* x    = (float*)(ws);                 // [2048,1024] f32   8 MB
  f16*   xln  = (f16*)(ws + 8388608);         // [2048,1024] f16   4 MB
  f16*   qkv  = (f16*)(ws + 12582912);        // [2048,3072] f16  12 MB
  f16*   obuf = (f16*)(ws + 25165824);        // [2048,1024] f16   4 MB
  f16*   hbuf = (f16*)(ws + 29360128);        // [2048,4096] f16  16 MB

  const int R = 2048;  // B*S rows

  k_embed<<<R, 256, 0, stream>>>(tokens, tok_emb, pos_emb, x);

  for (int l = 0; l < 8; l++) {
    k_ln<<<R, 256, 0, stream>>>(x, ln1_g + l * 1024, ln1_b + l * 1024, xln);
    k_gemm<0><<<dim3(24, 16), 256, 0, stream>>>(
        xln, attn_w + (size_t)l * 1024 * 3072, qkv, nullptr, R, 3072, 1024);
    k_attn<<<dim3(8, 16, 2), 256, 0, stream>>>(qkv, obuf);
    k_gemm<2><<<dim3(8, 16), 256, 0, stream>>>(
        obuf, attn_pw + (size_t)l * 1024 * 1024, nullptr, x, R, 1024, 1024);
    k_ln<<<R, 256, 0, stream>>>(x, ln2_g + l * 1024, ln2_b + l * 1024, xln);
    k_gemm<1><<<dim3(32, 16), 256, 0, stream>>>(
        xln, fc_w + (size_t)l * 1024 * 4096, hbuf, nullptr, R, 4096, 1024);
    k_gemm<2><<<dim3(8, 16), 256, 0, stream>>>(
        hbuf, proj_w + (size_t)l * 4096 * 1024, nullptr, x, R, 1024, 4096);
  }

  k_ln<<<R, 256, 0, stream>>>(x, lnf_g, lnf_b, xln);
  k_gemm<3><<<dim3(250, 16), 256, 0, stream>>>(xln, head_w, nullptr, out, R,
                                               32000, 1024);
}
